// GhostBotBlock_27187142983908
// MI455X (gfx1250) — matrix operand order repaired and checked by reading, not compiled, not run on hardware
//
#include <hip/hip_runtime.h>

// ---------------------------------------------------------------------------
// B=4 T=1024 D=768 H=8 HD=96 A=8 M=150.  Everything downstream of `summ` is
// t-invariant -> per-batch (4-row) compute, broadcast at the end.
// All GEMMs: bf16 operands; staging uses GLOBAL_LOAD_ASYNC_TO_LDS_B128 on the
// (uniform) full-tile fast path; f32 WMMA accumulation.
// ---------------------------------------------------------------------------
#define B_  4
#define T_  1024
#define D_  768
#define H_  8
#define HD_ 96
#define A_  8
#define M_  150
#define NT_ (B_*T_)

typedef __attribute__((ext_vector_type(16))) __bf16 v16bf;
typedef __attribute__((ext_vector_type(8)))  __bf16 v8bf;
typedef __attribute__((ext_vector_type(8)))  float  v8f;

__device__ inline unsigned short f2u(float x) {
  return (unsigned short)(__float_as_uint(x) >> 16);   // f32 -> bf16 (truncate)
}
__device__ inline unsigned int pk2(float a, float b) { // two bf16 packed
  return (__float_as_uint(a) >> 16) | (__float_as_uint(b) & 0xffff0000u);
}
__device__ inline float gelu_f(float x) {
  return 0.5f * x * (1.0f + erff(x * 0.7071067811865475f)); // exact gelu
}
__device__ inline float actf(float x, int a) {
  if (a == 1) return gelu_f(x);
  if (a == 2) return tanhf(x);
  if (a == 3) return 1.0f / (1.0f + expf(-x));
  return x;
}
__device__ inline v16bf cat8(v8bf lo, v8bf hi) {
  return __builtin_shufflevector(lo, hi, 0,1,2,3,4,5,6,7,8,9,10,11,12,13,14,15);
}
// CDNA5 async global->LDS copy of 16B per lane (IOFFSET applies to both sides).
__device__ inline void async_cp16(unsigned lds_off, const void* gsrc, int ioff) {
  switch (ioff) {
    case 0:  asm volatile("global_load_async_to_lds_b128 %0, %1, off"
                          :: "v"(lds_off), "v"(gsrc) : "memory"); break;
    case 16: asm volatile("global_load_async_to_lds_b128 %0, %1, off offset:16"
                          :: "v"(lds_off), "v"(gsrc) : "memory"); break;
    case 32: asm volatile("global_load_async_to_lds_b128 %0, %1, off offset:32"
                          :: "v"(lds_off), "v"(gsrc) : "memory"); break;
    default: asm volatile("global_load_async_to_lds_b128 %0, %1, off offset:48"
                          :: "v"(lds_off), "v"(gsrc) : "memory"); break;
  }
}
__device__ inline void async_wait0() {
  asm volatile("s_wait_asynccnt 0x0" ::: "memory");
}
__device__ inline unsigned lds_addr(const void* p) {   // generic LDS ptr -> offset
  return (unsigned)(size_t)p;
}

// ---------------------------------------------------------------------------
// f32 -> bf16 convert (8 elems/thread)
// ---------------------------------------------------------------------------
__global__ void cvt_bf16(const float* x, unsigned short* y, int n8) {
  int i = blockIdx.x * 256 + threadIdx.x;
  if (i >= n8) return;
  const float4* xp = (const float4*)x + (size_t)i * 2;
  float4 a = xp[0], b = xp[1];
  uint4 o;
  o.x = pk2(a.x, a.y); o.y = pk2(a.z, a.w);
  o.z = pk2(b.x, b.y); o.w = pk2(b.z, b.w);
  ((uint4*)y)[i] = o;
}

// f32 W[K,N] -> bf16 Wt[N,K] (tiled transpose-convert, paid once per weight)
__global__ void tcvt_bf16(const float* W, unsigned short* Wt, int K, int N) {
  __shared__ float tile[32][33];
  int n0 = blockIdx.x * 32, k0 = blockIdx.y * 32;
  int tx = threadIdx.x, ty = threadIdx.y;           // 32 x 8
  #pragma unroll
  for (int i = 0; i < 4; i++) {
    int k = k0 + ty + i * 8, n = n0 + tx;
    tile[ty + i * 8][tx] = (k < K && n < N) ? W[(size_t)k * N + n] : 0.f;
  }
  __syncthreads();
  #pragma unroll
  for (int i = 0; i < 4; i++) {
    int n = n0 + ty + i * 8, k = k0 + tx;
    if (n < N && k < K) Wt[(size_t)n * K + k] = f2u(tile[tx][ty + i * 8]);
  }
}

// f32 V (B,T,H,HD) -> bf16 Vt (B,H,HD,T)
__global__ void transpose_v(const float* V, unsigned short* Vt) {
  __shared__ float tile[32][33];
  int bh = blockIdx.z;
  int b = bh / H_, h = bh % H_;
  int hd0 = blockIdx.x * 32, t0 = blockIdx.y * 32;
  int tx = threadIdx.x, ty = threadIdx.y;
  #pragma unroll
  for (int i = 0; i < 4; i++) {
    int t = t0 + ty + i * 8, hd = hd0 + tx;
    tile[ty + i * 8][tx] = (hd < HD_) ? V[(size_t)(b * T_ + t) * D_ + h * HD_ + hd] : 0.f;
  }
  __syncthreads();
  #pragma unroll
  for (int i = 0; i < 4; i++) {
    int hd = hd0 + ty + i * 8, t = t0 + tx;
    if (hd < HD_) Vt[((size_t)bh * HD_ + hd) * T_ + t] = f2u(tile[tx][ty + i * 8]);
  }
}

// ---------------------------------------------------------------------------
// bf16-WMMA GEMM: C = act(A @ W' + bias); A bf16 [Mr,K], W bf16 TRANSPOSED
// [N,K]. 256 threads = 8 waves, 128x128 block tile, 32x64 per wave (8 WMMA /
// k-step). Full tiles staged with async-to-LDS; edge tiles with guarded copy.
// Requires K % 16 == 0.
// ---------------------------------------------------------------------------
struct GemmP {
  const unsigned short* A; const unsigned short* W; const float* bias; void* C;
  int Mr, N, K, ldc, colOff, act, outBf16;
  float* meanPtr; int meanLd, meanOff, rpb; float mscale;
};

#define GTM 128
#define GTN 128
#define GTK 32

__global__ __launch_bounds__(256) void gemm_wmma(GemmP p) {
  __shared__ unsigned short As[GTM * GTK];   // [row][k]
  __shared__ unsigned short Bs[GTN * GTK];   // [n][k]
  const int tid  = threadIdx.x;
  const int lane = tid & 31;
  const int wv   = tid >> 5;
  const int m0   = blockIdx.y * GTM;
  const int n0   = blockIdx.x * GTN;
  const int wm   = (wv & 3) * 32;
  const int wn   = (wv >> 2) * 64;
  const int fm   = lane & 15;
  const int klo  = (lane >> 4) * 8;    // C row offset
  const int kb16 = (lane >> 4) * 16;   // B frag K base
  const int cr   = tid >> 1, chf = tid & 1;  // staging: row/col, 16-elem half
  const bool fullM = (m0 + GTM) <= p.Mr;     // uniform
  const bool fullN = (n0 + GTN) <= p.N;      // uniform

  v8f acc[2][4];
  #pragma unroll
  for (int i = 0; i < 2; i++)
    #pragma unroll
    for (int j = 0; j < 4; j++)
      #pragma unroll
      for (int e = 0; e < 8; e++) acc[i][j][e] = 0.0f;

  for (int k0 = 0; k0 < p.K; k0 += GTK) {
    const bool kfull = (k0 + GTK) <= p.K;    // uniform
    { // ---- stage A
      int grow = m0 + cr;
      unsigned ldso = lds_addr(&As[cr * GTK + chf * 16]);
      const unsigned short* gsrc = p.A + (size_t)grow * p.K + k0 + chf * 16;
      if (fullM && kfull) {
        async_cp16(ldso, gsrc, 0);
        async_cp16(ldso, gsrc, 16);
      } else {
        uint4* dst = (uint4*)&As[cr * GTK + chf * 16];
        if (grow < p.Mr && (k0 + chf * 16 + 16) <= p.K) {
          const uint4* src = (const uint4*)gsrc;
          if (k0 + GTK < p.K) __builtin_prefetch(src + 2, 0, 1);
          dst[0] = src[0]; dst[1] = src[1];
        } else {
          uint4 z; z.x = z.y = z.z = z.w = 0u;
          dst[0] = z; dst[1] = z;
        }
      }
    }
    { // ---- stage B (W already [N,K] bf16)
      int gcol = n0 + cr;
      unsigned ldso = lds_addr(&Bs[cr * GTK + chf * 16]);
      const unsigned short* gsrc = p.W + (size_t)gcol * p.K + k0 + chf * 16;
      if (fullN && kfull) {
        async_cp16(ldso, gsrc, 0);
        async_cp16(ldso, gsrc, 16);
      } else {
        uint4* dst = (uint4*)&Bs[cr * GTK + chf * 16];
        if (gcol < p.N && (k0 + chf * 16 + 16) <= p.K) {
          const uint4* src = (const uint4*)gsrc;
          dst[0] = src[0]; dst[1] = src[1];
        } else {
          uint4 z; z.x = z.y = z.z = z.w = 0u;
          dst[0] = z; dst[1] = z;
        }
      }
    }
    async_wait0();
    __syncthreads();

    v16bf af[2], bf[4];
    #pragma unroll
    for (int i = 0; i < 2; i++) {
      const unsigned short* ar = &As[(wm + i * 16 + fm) * GTK];
      af[i] = cat8(*(const v8bf*)(ar + kb16), *(const v8bf*)(ar + kb16 + 8));
    }
    #pragma unroll
    for (int j = 0; j < 4; j++)
      bf[j] = *(const v16bf*)&Bs[(wn + j * 16 + fm) * GTK + kb16];
    #pragma unroll
    for (int i = 0; i < 2; i++)
      #pragma unroll
      for (int j = 0; j < 4; j++)
        acc[i][j] = __builtin_amdgcn_wmma_f32_16x16x32_bf16(
            false, af[i], false, bf[j], (short)0, acc[i][j], false, false);
    __syncthreads();
  }

  #pragma unroll
  for (int i = 0; i < 2; i++)
    #pragma unroll
    for (int j = 0; j < 4; j++) {
      int col = n0 + wn + j * 16 + fm;
      float bv = (p.bias != nullptr && col < p.N) ? p.bias[col] : 0.0f;
      #pragma unroll
      for (int e = 0; e < 8; e++) {
        int row = m0 + wm + i * 16 + klo + e;
        if (row < p.Mr && col < p.N) {
          float v = actf(acc[i][j][e] + bv, p.act);
          size_t ci = (size_t)row * p.ldc + p.colOff + col;
          if (p.outBf16) ((unsigned short*)p.C)[ci] = f2u(v);
          else           ((float*)p.C)[ci] = v;
          if (p.meanPtr) {
            int b = row / p.rpb;
            atomicAdd(p.meanPtr + (size_t)b * p.meanLd + p.meanOff + col, v * p.mscale);
          }
        }
      }
    }
}

// ---------------------------------------------------------------------------
// Flash attention: 4 waves/WG share one 32-key K/V LDS tile (async-staged);
// each wave owns a 16-row q block. bf16 in (Q,K row-major; V pre-transposed
// (B,H,HD,T)), bf16 out. 12 WMMA per key tile per wave.
// ---------------------------------------------------------------------------
__global__ __launch_bounds__(128) void flash_attn64(const unsigned short* Qb,
    const unsigned short* Kb, const unsigned short* Vt, unsigned short* Ob, float scale) {
  __shared__ unsigned short kt[32 * HD_];      // [key][hd]
  __shared__ unsigned short vt[HD_ * 32];      // [hd][key]
  __shared__ unsigned short pt[4 * 16 * 32];   // per-wave P tile
  const int tid  = threadIdx.x;
  const int lane = tid & 31;
  const int wv   = tid >> 5;
  const int ngq  = T_ / 64;
  const int bid  = blockIdx.x;
  const int gq   = bid % ngq;
  const int h    = (bid / ngq) % H_;
  const int b    = bid / (ngq * H_);
  const int q0   = gq * 64 + wv * 16;
  const int fm   = lane & 15;
  const int klo  = (lane >> 4) * 8;
  const int kb16 = (lane >> 4) * 16;
  unsigned short* ptw = &pt[wv * 16 * 32];

  v16bf qf[3];
  {
    const unsigned short* qrow = Qb + (size_t)(b * T_ + q0 + fm) * D_ + h * HD_;
    #pragma unroll
    for (int c = 0; c < 3; c++)
      qf[c] = cat8(*(const v8bf*)(qrow + c * 32 + kb16),
                   *(const v8bf*)(qrow + c * 32 + kb16 + 8));
  }
  float mrow[8], lrow[8];
  #pragma unroll
  for (int i = 0; i < 8; i++) { mrow[i] = -1e30f; lrow[i] = 0.0f; }
  v8f oacc[6];
  #pragma unroll
  for (int c = 0; c < 6; c++)
    #pragma unroll
    for (int e = 0; e < 8; e++) oacc[c][e] = 0.0f;

  for (int j = 0; j < T_ / 32; j++) {
    { // stage K rows: 128 threads x 24 hd, fully async
      int key = tid >> 2, seg = tid & 3;
      unsigned ldso = lds_addr(&kt[key * HD_ + seg * 24]);
      const unsigned short* gsrc = Kb + (size_t)(b * T_ + j * 32 + key) * D_ + h * HD_ + seg * 24;
      async_cp16(ldso, gsrc, 0);
      async_cp16(ldso, gsrc, 16);
      async_cp16(ldso, gsrc, 32);
    }
    if (tid < HD_) { // stage V^T rows (32 contiguous keys per hd row)
      unsigned ldso = lds_addr(&vt[tid * 32]);
      const unsigned short* gsrc = Vt + ((size_t)(b * H_ + h) * HD_ + tid) * T_ + j * 32;
      async_cp16(ldso, gsrc, 0);
      async_cp16(ldso, gsrc, 16);
      async_cp16(ldso, gsrc, 32);
      async_cp16(ldso, gsrc, 48);
    }
    async_wait0();
    __syncthreads();

    v8f s0, s1;
    #pragma unroll
    for (int e = 0; e < 8; e++) { s0[e] = 0.0f; s1[e] = 0.0f; }
    #pragma unroll
    for (int c = 0; c < 3; c++) {
      v16bf b0 = *(const v16bf*)&kt[fm * HD_ + c * 32 + kb16];
      v16bf b1 = *(const v16bf*)&kt[(16 + fm) * HD_ + c * 32 + kb16];
      s0 = __builtin_amdgcn_wmma_f32_16x16x32_bf16(false, qf[c], false, b0, (short)0, s0, false, false);
      s1 = __builtin_amdgcn_wmma_f32_16x16x32_bf16(false, qf[c], false, b1, (short)0, s1, false, false);
    }

    #pragma unroll
    for (int i = 0; i < 8; i++) {   // slot i -> q row i + 8*(lane>=16)
      float v0 = s0[i] * scale, v1 = s1[i] * scale;
      float mx = fmaxf(v0, v1);
      mx = fmaxf(mx, __shfl_xor(mx, 1, 32));
      mx = fmaxf(mx, __shfl_xor(mx, 2, 32));
      mx = fmaxf(mx, __shfl_xor(mx, 4, 32));
      mx = fmaxf(mx, __shfl_xor(mx, 8, 32));
      float mnew = fmaxf(mrow[i], mx);
      float p0 = expf(v0 - mnew), p1 = expf(v1 - mnew);
      float rs = p0 + p1;
      rs += __shfl_xor(rs, 1, 32);
      rs += __shfl_xor(rs, 2, 32);
      rs += __shfl_xor(rs, 4, 32);
      rs += __shfl_xor(rs, 8, 32);
      float alpha = expf(mrow[i] - mnew);
      lrow[i] = lrow[i] * alpha + rs;
      mrow[i] = mnew;
      #pragma unroll
      for (int c = 0; c < 6; c++) oacc[c][i] *= alpha;
      int prow = i + 8 * (lane >> 4);
      ptw[prow * 32 + fm]      = f2u(p0);
      ptw[prow * 32 + 16 + fm] = f2u(p1);
    }

    v16bf pa = cat8(*(const v8bf*)&ptw[fm * 32 + kb16],
                    *(const v8bf*)&ptw[fm * 32 + kb16 + 8]);
    #pragma unroll
    for (int c = 0; c < 6; c++) {
      v16bf vb = *(const v16bf*)&vt[(c * 16 + fm) * 32 + kb16];
      oacc[c] = __builtin_amdgcn_wmma_f32_16x16x32_bf16(false, pa, false, vb, (short)0, oacc[c], false, false);
    }
    __syncthreads();
  }

  #pragma unroll
  for (int c = 0; c < 6; c++)
    #pragma unroll
    for (int e = 0; e < 8; e++) {
      int t = q0 + klo + e;
      Ob[(size_t)(b * T_ + t) * D_ + h * HD_ + c * 16 + fm] = f2u(oacc[c][e] / lrow[e]);
    }
}

// ---------------------------------------------------------------------------
// Per-batch memory attention (q t-invariant): 1 wave per batch; bf16 out.
// ---------------------------------------------------------------------------
__global__ __launch_bounds__(32) void mem_attn(const float* q, const float* k,
                                               const float* v, unsigned short* outp, float scale) {
  __shared__ float qs[D_];
  __shared__ float sc[M_];
  const int lane = threadIdx.x;
  const int b = blockIdx.x;
  for (int d = lane; d < D_; d += 32) qs[d] = q[b * D_ + d];
  __syncthreads();
  for (int m = lane; m < M_; m += 32) {
    float acc = 0.f;
    const float* kr = k + ((size_t)(b * M_ + m)) * D_;
    for (int d = 0; d < D_; d++) acc += qs[d] * kr[d];
    sc[m] = acc * scale;
  }
  __syncthreads();
  float mx = -1e30f;
  for (int m = lane; m < M_; m += 32) mx = fmaxf(mx, sc[m]);
  for (int msk = 16; msk > 0; msk >>= 1) mx = fmaxf(mx, __shfl_xor(mx, msk, 32));
  float s = 0.f;
  for (int m = lane; m < M_; m += 32) { float e = expf(sc[m] - mx); sc[m] = e; s += e; }
  for (int msk = 16; msk > 0; msk >>= 1) s += __shfl_xor(s, msk, 32);
  __syncthreads();
  float inv = 1.0f / s;
  float accv[D_ / 32];
  #pragma unroll
  for (int i = 0; i < D_ / 32; i++) accv[i] = 0.f;
  for (int m = 0; m < M_; m++) {
    float w = sc[m] * inv;
    const float* vr = v + ((size_t)(b * M_ + m)) * D_;
    #pragma unroll
    for (int i = 0; i < D_ / 32; i++) accv[i] += w * vr[lane + 32 * i];
  }
  #pragma unroll
  for (int i = 0; i < D_ / 32; i++) outp[b * D_ + lane + 32 * i] = f2u(accv[i]);
}

// ---------------------------------------------------------------------------
// LayerNorm (optional residual, optional post-gelu, optional bf16 copy out)
// ---------------------------------------------------------------------------
__global__ __launch_bounds__(256) void ln_vec(const float* x, const float* res,
                                              const float* g, const float* bb,
                                              float* out, unsigned short* outb,
                                              int L, int postGelu) {
  __shared__ float red[256];
  const int row = blockIdx.x, tid = threadIdx.x;
  const float* xr = x + (size_t)row * L;
  const float* rr = res ? res + (size_t)row * L : nullptr;
  float s = 0.f;
  for (int d = tid; d < L; d += 256) { float v = xr[d] + (rr ? rr[d] : 0.f); s += v; }
  red[tid] = s; __syncthreads();
  for (int o = 128; o > 0; o >>= 1) { if (tid < o) red[tid] += red[tid + o]; __syncthreads(); }
  float mean = red[0] / (float)L;
  __syncthreads();
  float sq = 0.f;
  for (int d = tid; d < L; d += 256) { float v = xr[d] + (rr ? rr[d] : 0.f) - mean; sq += v * v; }
  red[tid] = sq; __syncthreads();
  for (int o = 128; o > 0; o >>= 1) { if (tid < o) red[tid] += red[tid + o]; __syncthreads(); }
  float rstd = rsqrtf(red[0] / (float)L + 1e-5f);
  __syncthreads();
  for (int d = tid; d < L; d += 256) {
    float v = xr[d] + (rr ? rr[d] : 0.f);
    float y = (v - mean) * rstd * g[d] + bb[d];
    if (postGelu) y = gelu_f(y);
    if (out)  out[(size_t)row * L + d] = y;
    if (outb) outb[(size_t)row * L + d] = f2u(y);
  }
}

__global__ __launch_bounds__(256) void softmax_row(float* x, int L) {
  __shared__ float red[256];
  const int row = blockIdx.x, tid = threadIdx.x;
  float* xr = x + (size_t)row * L;
  float mx = -1e30f;
  for (int d = tid; d < L; d += 256) mx = fmaxf(mx, xr[d]);
  red[tid] = mx; __syncthreads();
  for (int o = 128; o > 0; o >>= 1) { if (tid < o) red[tid] = fmaxf(red[tid], red[tid + o]); __syncthreads(); }
  mx = red[0]; __syncthreads();
  float s = 0.f;
  for (int d = tid; d < L; d += 256) { float e = expf(xr[d] - mx); xr[d] = e; s += e; }
  red[tid] = s; __syncthreads();
  for (int o = 128; o > 0; o >>= 1) { if (tid < o) red[tid] += red[tid + o]; __syncthreads(); }
  float inv = 1.0f / red[0]; __syncthreads();
  for (int d = tid; d < L; d += 256) xr[d] *= inv;
}

__global__ void mean_time_k(const float* x, float* summ, int colOff) {
  int idx = blockIdx.x * 256 + threadIdx.x;            // float4 index
  if (idx >= B_ * D_ / 4) return;
  int b = idx / (D_ / 4), d4 = idx % (D_ / 4);
  float4 s; s.x = s.y = s.z = s.w = 0.f;
  const float4* xp = (const float4*)x;
  for (int t = 0; t < T_; t++) {
    float4 v = xp[(size_t)(b * T_ + t) * (D_ / 4) + d4];
    s.x += v.x; s.y += v.y; s.z += v.z; s.w += v.w;
  }
  const float inv = 1.0f / (float)T_;
  s.x *= inv; s.y *= inv; s.z *= inv; s.w *= inv;
  ((float4*)summ)[((size_t)b * (7 * D_) + colOff) / 4 + d4] = s;
}

__global__ void zero_f(float* p, int n) { int i = blockIdx.x * 256 + threadIdx.x; if (i < n) p[i] = 0.f; }

__global__ void scale_mem(const float* mem, const float* decayP, float* memd,
                          unsigned short* memdb, int n4) {
  int i = blockIdx.x * 256 + threadIdx.x;
  if (i >= n4) return;
  float dly = decayP[0];
  float4 v = ((const float4*)mem)[i];
  v.x *= dly; v.y *= dly; v.z *= dly; v.w *= dly;
  ((float4*)memd)[i] = v;
  uint2 o; o.x = pk2(v.x, v.y); o.y = pk2(v.z, v.w);
  ((uint2*)memdb)[i] = o;
}

__global__ void mul_cvt(float* y, const float* g, unsigned short* yb, int n) {
  int i = blockIdx.x * 256 + threadIdx.x;
  if (i >= n) return;
  float v = y[i] * g[i];
  y[i] = v; yb[i] = f2u(v);
}

__global__ void bcast_out(const float* v, float* out) {
  int i = blockIdx.x * 256 + threadIdx.x;              // float4 index
  if (i >= NT_ * D_ / 4) return;
  int b = i / (T_ * D_ / 4);
  int d4 = i % (D_ / 4);
  ((float4*)out)[i] = ((const float4*)v)[b * (D_ / 4) + d4];
}

__global__ void update_mem(const float* memd, const float* ww, const float* cont, float* outm) {
  int i = blockIdx.x * 256 + threadIdx.x;              // float4 index
  if (i >= B_ * M_ * D_ / 4) return;
  int b = i / (M_ * D_ / 4);
  int r = i % (M_ * D_ / 4);
  int m = r / (D_ / 4), d4 = r % (D_ / 4);
  float w = ww[b * M_ + m];
  float4 md = ((const float4*)memd)[i];
  float4 c  = ((const float4*)cont)[b * (D_ / 4) + d4];
  md.x += w * c.x; md.y += w * c.y; md.z += w * c.z; md.w += w * c.w;
  ((float4*)outm)[i] = md;
}

// GRU over T steps; 32 lanes = (b=lane/8, a=lane%8)
__global__ __launch_bounds__(32) void gru_scan(const float* emoin, const float* prev,
    const float* wih, const float* bih, const float* whh, const float* bhh,
    const float* baseline, const float* regP, float* uni, float* fin) {
  const int lane = threadIdx.x;
  const int b = lane >> 3, a = lane & 7;
  float wI[8][3], wH[8][3], bI[3], bH[3];
  #pragma unroll
  for (int g = 0; g < 3; g++) { bI[g] = bih[g * 8 + a]; bH[g] = bhh[g * 8 + a]; }
  #pragma unroll
  for (int j = 0; j < 8; j++)
    #pragma unroll
    for (int g = 0; g < 3; g++) { wI[j][g] = wih[j * 24 + g * 8 + a]; wH[j][g] = whh[j * 24 + g * 8 + a]; }
  float h = prev[b * 8 + a];
  float reg = regP[0];
  float base = baseline[a];
  int grp = lane & 24;
  for (int t = 0; t < T_; t++) {
    float xo = emoin[((size_t)(b * T_ + t)) * 8 + a];
    float gi0 = bI[0], gi1 = bI[1], gi2 = bI[2], gh0 = bH[0], gh1 = bH[1], gh2 = bH[2];
    #pragma unroll
    for (int j = 0; j < 8; j++) {
      float xj = __shfl(xo, grp | j, 32);
      float hj = __shfl(h,  grp | j, 32);
      gi0 += xj * wI[j][0]; gi1 += xj * wI[j][1]; gi2 += xj * wI[j][2];
      gh0 += hj * wH[j][0]; gh1 += hj * wH[j][1]; gh2 += hj * wH[j][2];
    }
    float r = 1.f / (1.f + expf(-(gi0 + gh0)));
    float z = 1.f / (1.f + expf(-(gi1 + gh1)));
    float n = tanhf(gi2 + r * gh2);
    float hn = (1.f - z) * n + z * h;
    hn = hn + reg * (base - hn);
    hn = tanhf(hn);
    h = hn;
    uni[((size_t)(b * T_ + t)) * 8 + a] = h;
  }
  fin[b * 8 + a] = h;
}

// ---------------------------------------------------------------------------
// Host orchestration
// ---------------------------------------------------------------------------
extern "C" void kernel_launch(void* const* d_in, const int* in_sizes, int n_in,
                              void* d_out, int out_size, void* d_ws, size_t ws_size,
                              hipStream_t stream) {
  (void)in_sizes; (void)n_in; (void)out_size; (void)ws_size;
  #define PIN(i) ((const float*)d_in[(i)])
  const float* vis  = PIN(0);
  const float* aud  = PIN(1);
  const float* lang = PIN(2);
  const float* touch= PIN(3);
  const float* prop = PIN(4);
  const float* vest = PIN(5);
  const float* bci  = PIN(6);
  const float* mem  = PIN(7);
  const float* prev = PIN(8);
  const int P0 = 9;

  float* outp    = (float*)d_out;
  float* out_mem = outp + (size_t)NT_ * D_;
  float* out_uni = out_mem + (size_t)B_ * M_ * D_;
  float* out_fin = out_uni + (size_t)NT_ * A_;

  float* wsf = (float*)d_ws;
  size_t wo = 0;
  auto alloc_f = [&](size_t n) { float* p = wsf + wo; wo += (n + 7) & ~(size_t)7; return p; };
  auto alloc_h = [&](size_t n) { return (unsigned short*)alloc_f((n + 1) / 2); };

  unsigned short* visb  = alloc_h((size_t)NT_ * D_);
  unsigned short* audb  = alloc_h((size_t)NT_ * D_);
  unsigned short* touchb= alloc_h((size_t)NT_ * D_);
  unsigned short* bcib  = alloc_h((size_t)NT_ * D_);
  unsigned short* qbf   = alloc_h((size_t)NT_ * D_);
  unsigned short* kbf   = alloc_h((size_t)NT_ * D_);
  unsigned short* vtg   = alloc_h((size_t)B_ * H_ * HD_ * T_);
  unsigned short* aobf  = alloc_h((size_t)NT_ * D_);
  unsigned short* hhb   = alloc_h((size_t)NT_ * (D_ / 2));
  unsigned short* catb  = alloc_h((size_t)NT_ * (4 * A_));
  unsigned short* h16b  = alloc_h((size_t)NT_ * (2 * A_));
  unsigned short* memdb = alloc_h((size_t)B_ * M_ * D_);
  unsigned short* summb = alloc_h((size_t)B_ * 7 * D_);
  unsigned short* hgelub= alloc_h((size_t)B_ * 2 * D_);
  unsigned short* fusedvb = alloc_h((size_t)B_ * D_);
  unsigned short* retpb = alloc_h((size_t)B_ * D_);
  unsigned short* f2b   = alloc_h((size_t)B_ * D_);
  unsigned short* ffhb  = alloc_h((size_t)B_ * 4 * D_);
  unsigned short* wtb   = alloc_h((size_t)(7 * D_) * (2 * D_));  // max weight (5376x1536)
  float* vb    = alloc_f((size_t)NT_ * D_);
  float* emoin = alloc_f((size_t)NT_ * A_);
  float* summ  = alloc_f((size_t)B_ * 7 * D_);
  float* gate  = alloc_f((size_t)B_ * D_);
  float* hfuse = alloc_f((size_t)B_ * 2 * D_);
  float* hgelu = alloc_f((size_t)B_ * 2 * D_);
  float* fusedv= alloc_f((size_t)B_ * D_);
  float* memd  = alloc_f((size_t)B_ * M_ * D_);
  float* mk    = alloc_f((size_t)B_ * M_ * D_);
  float* mv    = alloc_f((size_t)B_ * M_ * D_);
  float* mq    = alloc_f((size_t)B_ * D_);
  float* retr  = alloc_f((size_t)B_ * D_);
  float* wwb   = alloc_f((size_t)B_ * M_);
  float* cont  = alloc_f((size_t)B_ * D_);
  float* f2    = alloc_f((size_t)B_ * D_);
  float* ffo   = alloc_f((size_t)B_ * D_);
  float* outv  = alloc_f((size_t)B_ * D_);

  const float scMha = 1.0f / sqrtf((float)HD_);
  const float scMem = 1.0f / sqrtf((float)D_);
  const int SLD = 7 * D_;
  const float invT = 1.0f / (float)T_;

  auto cvt = [&](const float* x, unsigned short* y, size_t n) {
    cvt_bf16<<<((n / 8) + 255) / 256, 256, 0, stream>>>(x, y, (int)(n / 8));
  };
  auto gemm = [&](const unsigned short* Abf, const float* Wf, int K, int N,
                  const float* bias, void* C, int Mr, int ldc, int colOff,
                  int act, int outBf16, float* meanPtr = nullptr, int meanOff = 0) {
    dim3 tg((N + 31) / 32, (K + 31) / 32);
    tcvt_bf16<<<tg, dim3(32, 8), 0, stream>>>(Wf, wtb, K, N);
    GemmP p; p.A = Abf; p.W = wtb; p.bias = bias; p.C = C;
    p.Mr = Mr; p.N = N; p.K = K; p.ldc = ldc; p.colOff = colOff; p.act = act;
    p.outBf16 = outBf16;
    p.meanPtr = meanPtr; p.meanLd = SLD; p.meanOff = meanOff; p.rpb = T_; p.mscale = invT;
    dim3 g((N + GTN - 1) / GTN, (Mr + GTM - 1) / GTM);
    gemm_wmma<<<g, 256, 0, stream>>>(p);
  };

  cvt(vis,   visb,   (size_t)NT_ * D_);
  cvt(aud,   audb,   (size_t)NT_ * D_);
  cvt(touch, touchb, (size_t)NT_ * D_);
  cvt(bci,   bcib,   (size_t)NT_ * D_);
  zero_f<<<(B_ * SLD + 255) / 256, 256, 0, stream>>>(summ, B_ * SLD);

  dim3 tvg(HD_ / 32, T_ / 32, B_ * H_);
  // 'va': aud_enh = MHA(q=aud, kv=vis) -> mean into summ[768:1536]
  gemm(audb, PIN(P0+0), D_, D_, PIN(P0+1), qbf, NT_, D_, 0, 0, 1);
  gemm(visb, PIN(P0+2), D_, D_, PIN(P0+3), kbf, NT_, D_, 0, 0, 1);
  gemm(visb, PIN(P0+4), D_, D_, PIN(P0+5), vb,  NT_, D_, 0, 0, 0);
  transpose_v<<<tvg, dim3(32, 8), 0, stream>>>(vb, vtg);
  flash_attn64<<<B_ * H_ * (T_ / 64), 128, 0, stream>>>(qbf, kbf, vtg, aobf, scMha);
  gemm(aobf, PIN(P0+6), D_, D_, PIN(P0+7), vb, NT_, D_, 0, 0, 0, summ, D_);

  // 'av': vis_enh = MHA(q=vis, kv=aud) -> mean into summ[0:768]
  gemm(visb, PIN(P0+8),  D_, D_, PIN(P0+9),  qbf, NT_, D_, 0, 0, 1);
  gemm(audb, PIN(P0+10), D_, D_, PIN(P0+11), kbf, NT_, D_, 0, 0, 1);
  gemm(audb, PIN(P0+12), D_, D_, PIN(P0+13), vb,  NT_, D_, 0, 0, 0);
  transpose_v<<<tvg, dim3(32, 8), 0, stream>>>(vb, vtg);
  flash_attn64<<<B_ * H_ * (T_ / 64), 128, 0, stream>>>(qbf, kbf, vtg, aobf, scMha);
  gemm(aobf, PIN(P0+14), D_, D_, PIN(P0+15), vb, NT_, D_, 0, 0, 0, summ, 0);

  dim3 mgrid((B_ * D_ / 4 + 255) / 256);
  mean_time_k<<<mgrid, 256, 0, stream>>>(lang,  summ, 2 * D_);
  mean_time_k<<<mgrid, 256, 0, stream>>>(touch, summ, 3 * D_);
  mean_time_k<<<mgrid, 256, 0, stream>>>(prop,  summ, 4 * D_);
  mean_time_k<<<mgrid, 256, 0, stream>>>(vest,  summ, 5 * D_);
  mean_time_k<<<mgrid, 256, 0, stream>>>(bci,   summ, 6 * D_);
  cvt(summ, summb, (size_t)B_ * SLD);

  gemm(summb, PIN(25), 7 * D_, D_,     PIN(26), gate,  B_, D_,     0, 3, 0);
  gemm(summb, PIN(27), 7 * D_, 2 * D_, PIN(28), hfuse, B_, 2 * D_, 0, 0, 0);
  ln_vec<<<B_, 256, 0, stream>>>(hfuse, nullptr, PIN(29), PIN(30), hgelu, hgelub, 2 * D_, 1);
  gemm(hgelub, PIN(31), 2 * D_, D_, PIN(32), fusedv, B_, D_, 0, 0, 0);
  mul_cvt<<<(B_ * D_ + 255) / 256, 256, 0, stream>>>(fusedv, gate, fusedvb, B_ * D_);

  scale_mem<<<(B_ * M_ * D_ / 4 + 255) / 256, 256, 0, stream>>>(mem, PIN(43), memd, memdb, B_ * M_ * D_ / 4);
  gemm(memdb,   PIN(35), D_, D_, PIN(36), mk, B_ * M_, D_, 0, 0, 0);
  gemm(memdb,   PIN(37), D_, D_, PIN(38), mv, B_ * M_, D_, 0, 0, 0);
  gemm(fusedvb, PIN(33), D_, D_, PIN(34), mq, B_,      D_, 0, 0, 0);
  mem_attn<<<B_, 32, 0, stream>>>(mq, mk, mv, retpb, scMem);
  gemm(retpb,   PIN(39), D_, D_, PIN(40), retr, B_, D_, 0, 0, 0);
  gemm(fusedvb, PIN(44), D_, M_, PIN(45), wwb,  B_, M_, 0, 0, 0);
  softmax_row<<<B_, 256, 0, stream>>>(wwb, M_);
  gemm(fusedvb, PIN(41), D_, D_, PIN(42), cont, B_, D_, 0, 0, 0);
  update_mem<<<(B_ * M_ * D_ / 4 + 255) / 256, 256, 0, stream>>>(memd, wwb, cont, out_mem);

  ln_vec<<<B_, 256, 0, stream>>>(retr, fusedv, PIN(76), PIN(77), f2, f2b, D_, 0);
  gemm(f2b,  PIN(72), D_,     4 * D_, PIN(73), ffhb, B_, 4 * D_, 0, 1, 1);
  gemm(ffhb, PIN(74), 4 * D_, D_,     PIN(75), ffo,  B_, D_,     0, 0, 0);
  ln_vec<<<B_, 256, 0, stream>>>(ffo, f2, PIN(78), PIN(79), outv, nullptr, D_, 0);
  bcast_out<<<((size_t)NT_ * D_ / 4 + 255) / 256, 256, 0, stream>>>(outv, outp);

  const unsigned short* bx[4] = { visb, audb, touchb, bcib };
  const int bpi[4] = { 46, 50, 54, 58 };
  for (int i = 0; i < 4; i++) {
    gemm(bx[i], PIN(bpi[i]+0), D_,     D_ / 2, PIN(bpi[i]+1), hhb,  NT_, D_ / 2, 0,       1, 1);
    gemm(hhb,   PIN(bpi[i]+2), D_ / 2, A_,     PIN(bpi[i]+3), catb, NT_, 4 * A_, A_ * i,  2, 1);
  }
  gemm(catb, PIN(62), 4 * A_, 2 * A_, PIN(63), h16b,  NT_, 2 * A_, 0, 1, 1);
  gemm(h16b, PIN(64), 2 * A_, A_,     PIN(65), emoin, NT_, A_,     0, 2, 0);
  gru_scan<<<1, 32, 0, stream>>>(emoin, prev, PIN(66), PIN(67), PIN(68), PIN(69),
                                 PIN(70), PIN(71), out_uni, out_fin);
  #undef PIN
}
